// Combined_Geo_Encoding_Volume_48129403519654
// MI455X (gfx1250) — matrix operand order repaired and checked by reading, not compiled, not run on hardware
//
#include <hip/hip_runtime.h>
#include <math.h>

// CDNA5 (gfx1250) wave32 WMMA implementation.
typedef __attribute__((ext_vector_type(16))) _Float16 v16h;
typedef __attribute__((ext_vector_type(8)))  _Float16 v8h;
typedef __attribute__((ext_vector_type(8)))  float    v8f;

#define B_   4
#define C_   128
#define H_   128
#define W_   256
#define F_   128
#define G_   96
#define CV_  49
#define HW_  (H_ * W_)
#define NCORR ((size_t)B_ * CV_ * H_ * W_)

// workspace layout (bytes)
#define F1OFF ((size_t)0)
#define F2OFF ((size_t)B_ * H_ * W_ * F_ * 2)   // 33.55 MB of f16 fmap1n
#define MOFF  (2 * F2OFF)                        // folded selector matrices (444 f32)
#define WOFF  (MOFF + 4096)                      // f16 weights [2][128][128]

static __device__ __forceinline__ void wait_async0() {
#if __has_builtin(__builtin_amdgcn_s_wait_asynccnt)
    __builtin_amdgcn_s_wait_asynccnt(0);
#else
    asm volatile("s_wait_asynccnt 0x0" ::: "memory");
#endif
}

// -------------------------------------------------------------------------
// One-shot: convert W_f1 / W_f2 (128x128 f32, row-major [o][c]) to f16.
// -------------------------------------------------------------------------
__global__ __launch_bounds__(256)
void wconv_kernel(const float* __restrict__ Wf1, const float* __restrict__ Wf2,
                  _Float16* __restrict__ Wh)
{
    const int i = blockIdx.x * 256 + threadIdx.x;     // 0..32767
    const float* src = (i < F_ * C_) ? Wf1 : Wf2;
    Wh[i] = (_Float16)src[i & (F_ * C_ - 1)];
}

// -------------------------------------------------------------------------
// Kernel 1: fmap = W (128x128) @ feat + b, channel-L2 normalize, store f16
// in [b,h,w,c] layout. feat tile is DMA'd into LDS with
// global_load_async_to_lds_b128 (raw f32 [c][w]); output goes through an
// LDS f16 tile so global stores are consecutive-lane b128 chunks.
// -------------------------------------------------------------------------
__global__ __launch_bounds__(128)
void fmap_kernel(const float* __restrict__ featL, const float* __restrict__ featR,
                 const _Float16* __restrict__ Whalf,
                 const float* __restrict__ bf1, const float* __restrict__ bf2,
                 _Float16* __restrict__ out1, _Float16* __restrict__ out2)
{
    __shared__ float    sF[C_ * 16];      // [c][w] f32 tile (8 KB), async-DMA target
    __shared__ _Float16 sOut[16 * F_];    // [w][c] normalized f16 tile (4 KB)
    __shared__ float    sNorm[16];

    const int t    = threadIdx.x;
    const int lane = t & 31;
    const int wave = t >> 5;
    const int m    = lane & 15;   // A row / B column / C column index
    const int g    = lane >> 4;   // half-wave group

    const int w0 = blockIdx.x * 16;
    const int h  = blockIdx.y;
    const int s  = blockIdx.z & 1;
    const int b  = blockIdx.z >> 1;

    const float*    feat = s ? featR : featL;
    const _Float16* Wm   = Whalf + (size_t)s * (F_ * C_);
    const float*    bias = s ? bf2 : bf1;
    _Float16*       outp = s ? out2 : out1;

    if (t < 16) sNorm[t] = 0.0f;

    // Async DMA: each lane copies 16B (4 f32 of one channel row) into LDS.
#pragma unroll
    for (int i = 0; i < 4; ++i) {
        const int idx   = t + 128 * i;       // (c, 4-float chunk)
        const int c     = idx >> 2;
        const int chunk = idx & 3;
        const unsigned goff = (unsigned)(((((unsigned)b * C_ + c) * HW_)
                              + (unsigned)h * W_ + w0 + chunk * 4) * sizeof(float));
        const unsigned loff = (unsigned)(uintptr_t)(sF + c * 16 + chunk * 4);
        asm volatile("global_load_async_to_lds_b128 %0, %1, %2"
                     :: "v"(loff), "v"(goff), "s"(feat) : "memory");
    }
    wait_async0();
    __syncthreads();

    v8f acc0, acc1;
#pragma unroll
    for (int r = 0; r < 8; ++r) { acc0[r] = 0.0f; acc1[r] = 0.0f; }

#pragma unroll
    for (int kk = 0; kk < 4; ++kk) {
        const int c0 = kk * 32;
        // B fragment (32x16): element e -> K=c0+8g+e (e<8), c0+16+8g+(e-8) (e>=8).
        // Strided LDS reads: 16 lanes hit 16 consecutive banks (conflict-free).
        v16h bf;
#pragma unroll
        for (int e = 0; e < 16; ++e)
            bf[e] = (_Float16)sF[(c0 + g * 8 + e + (e >= 8 ? 8 : 0)) * 16 + m];
#pragma unroll
        for (int ti = 0; ti < 2; ++ti) {
            const _Float16* ap = Wm + (size_t)(wave * 32 + ti * 16 + m) * C_ + c0;
            const v8h alo = *(const v8h*)(ap + g * 8);
            const v8h ahi = *(const v8h*)(ap + 16 + g * 8);
            const v16h a = __builtin_shufflevector(alo, ahi,
                              0,1,2,3,4,5,6,7,8,9,10,11,12,13,14,15);
            if (ti == 0)
                acc0 = __builtin_amdgcn_wmma_f32_16x16x32_f16(false, a, false, bf,
                                                              (short)0, acc0, false, false);
            else
                acc1 = __builtin_amdgcn_wmma_f32_16x16x32_f16(false, a, false, bf,
                                                              (short)0, acc1, false, false);
        }
    }

    // bias + sum-of-squares (all 16 values of this thread share pixel w0+m)
    float fv[16];
    float ss = 0.0f;
#pragma unroll
    for (int r = 0; r < 8; ++r) {
        const float v0 = acc0[r] + bias[wave * 32 + 0 * 16 + 8 * g + r];
        const float v1 = acc1[r] + bias[wave * 32 + 1 * 16 + 8 * g + r];
        fv[r] = v0; fv[8 + r] = v1;
        ss += v0 * v0 + v1 * v1;
    }
    atomicAdd(&sNorm[m], ss);        // ds_add_f32
    __syncthreads();
    const float inv = 1.0f / (sqrtf(sNorm[m]) + 1e-8f);

    // Stage normalized f16 into LDS [w][c], then write tile coalesced:
    // the 16x128 f16 tile is contiguous in the [b,h,w,c] global layout.
#pragma unroll
    for (int ti = 0; ti < 2; ++ti) {
        v8h pk;
#pragma unroll
        for (int r = 0; r < 8; ++r) pk[r] = (_Float16)(fv[ti * 8 + r] * inv);
        *(v8h*)(sOut + m * F_ + wave * 32 + ti * 16 + 8 * g) = pk;
    }
    __syncthreads();

    _Float16* tileBase = outp + (((size_t)b * H_ + h) * W_ + w0) * F_;  // 4 KB contiguous
#pragma unroll
    for (int i = 0; i < 2; ++i) {
        const int chunk = t + 128 * i;              // 256 chunks of 8 halves (16B)
        *(v8h*)(tileBase + chunk * 8) = *(const v8h*)(sOut + chunk * 8);
    }
}

// -------------------------------------------------------------------------
// Kernel 2: banded correlation. Per (b,h) row, per 16-pixel x-tile compute
// C[x, x2] = f1n[x]·f2n[x2] for the 5 overlapping x2-tiles via WMMA.
// Accumulators are transposed through LDS [x_local][k] so init_corr is
// written with fully coalesced consecutive-lane stores.
// -------------------------------------------------------------------------
__global__ __launch_bounds__(128)
void corr_kernel(const _Float16* __restrict__ f1n, const _Float16* __restrict__ f2n,
                 float* __restrict__ corr)
{
    __shared__ float sX[64 * 53];   // [x_local 0..63][k 0..48], pad 53 (odd stride)

    const int t    = threadIdx.x;
    const int lane = t & 31;
    const int wave = t >> 5;
    const int m    = lane & 15;
    const int g    = lane >> 4;

    const int x0 = (blockIdx.x * 4 + wave) * 16;
    const int h  = blockIdx.y;
    const int b  = blockIdx.z;

    const _Float16* r1 = f1n + ((size_t)b * H_ + h) * (size_t)W_ * F_;
    const _Float16* r2 = f2n + ((size_t)b * H_ + h) * (size_t)W_ * F_;

    v8f acc[5];
#pragma unroll
    for (int j = 0; j < 5; ++j)
#pragma unroll
        for (int r = 0; r < 8; ++r) acc[j][r] = 0.0f;

#pragma unroll
    for (int kk = 0; kk < 4; ++kk) {
        const int c0 = kk * 32;
        const _Float16* ap = r1 + (size_t)(x0 + m) * F_ + c0;
        const v16h a = *(const v16h*)(ap + g * 16);
#pragma unroll
        for (int j = 0; j < 5; ++j) {
            const int x2 = x0 + (j - 2) * 16 + m;
            v16h bf;
            if ((unsigned)x2 < (unsigned)W_) {
                bf = *(const v16h*)(r2 + (size_t)x2 * F_ + c0 + g * 16);
            } else {
#pragma unroll
                for (int q = 0; q < 16; ++q) bf[q] = (_Float16)0.0f;
            }
            acc[j] = __builtin_amdgcn_wmma_f32_16x16x32_f16(false, a, false, bf,
                                                            (short)0, acc[j], false, false);
        }
    }

    // Transpose band into LDS: C lane -> (M = r + 8g, N = m); k = 24+M-N-16(j-2)
#pragma unroll
    for (int j = 0; j < 5; ++j)
#pragma unroll
        for (int r = 0; r < 8; ++r) {
            const int M = r + 8 * g;
            const int k = 24 + M - m - 16 * (j - 2);
            if (0 <= k && k < CV_)
                sX[(wave * 16 + M) * 53 + k] = acc[j][r];
        }
    __syncthreads();

    // Coalesced writeback: 49 rows of 64 consecutive pixels.
    const int x0wg = blockIdx.x * 64;
#pragma unroll
    for (int i = 0; i < 25; ++i) {
        const int idx = t + 128 * i;          // 0..3135 = (k, xl)
        if (idx < CV_ * 64) {
            const int k  = idx >> 6;
            const int xl = idx & 63;
            corr[(((size_t)b * CV_ + k) * H_ + h) * W_ + x0wg + xl] = sX[xl * 53 + k];
        }
    }
}

// -------------------------------------------------------------------------
// Fold selector: M[j][i][k] = sum_g Wsel[j, i*96+g] * Wgeo[g, k],
// c[j] = bsel[j] + sum_{i,g} Wsel[j,i*96+g]*bgeo[g].  (tiny, one block)
// -------------------------------------------------------------------------
__global__ void fold_kernel(const float* __restrict__ Wsel, const float* __restrict__ bsel,
                            const float* __restrict__ Wgeo, const float* __restrict__ bgeo,
                            float* __restrict__ Mf)
{
    const int t = threadIdx.x;
    if (t < 441) {
        const int j = t / 147;
        const int rem = t % 147;
        const int i = rem / 49;
        const int k = rem % 49;
        float s = 0.0f;
        for (int gg = 0; gg < G_; ++gg)
            s += Wsel[j * (3 * G_) + i * G_ + gg] * Wgeo[gg * CV_ + k];
        Mf[(j * 3 + i) * CV_ + k] = s;
    } else if (t < 444) {
        const int j = t - 441;
        float s = bsel[j];
        for (int i = 0; i < 3; ++i)
            for (int gg = 0; gg < G_; ++gg)
                s += Wsel[j * (3 * G_) + i * G_ + gg] * bgeo[gg];
        Mf[441 + j] = s;
    }
}

// -------------------------------------------------------------------------
// Kernel 3: streaming selection. logits via folded 3x49 matrices, softmax,
// final = init_corr + w0*cv0 + w1*cv1 + w2*cv2.  Pure bandwidth.
// -------------------------------------------------------------------------
__global__ __launch_bounds__(256)
void select_kernel(const float* __restrict__ cv0, const float* __restrict__ cv1,
                   const float* __restrict__ cv2, const float* __restrict__ Mf,
                   const float* __restrict__ corr, float* __restrict__ final_out,
                   float* __restrict__ w_out)
{
    const size_t p  = (size_t)blockIdx.x * blockDim.x + threadIdx.x;
    const size_t b  = p / HW_;
    const size_t hw = p % HW_;
    const size_t base = b * (size_t)CV_ * HW_ + hw;

    float l0 = Mf[441 + 0], l1 = Mf[441 + 1], l2 = Mf[441 + 2];
#pragma unroll 7
    for (int k = 0; k < CV_; ++k) {
        const size_t off = base + (size_t)k * HW_;
        const float v0 = cv0[off], v1 = cv1[off], v2 = cv2[off];
        l0 += Mf[(0 * 3 + 0) * CV_ + k] * v0 + Mf[(0 * 3 + 1) * CV_ + k] * v1 + Mf[(0 * 3 + 2) * CV_ + k] * v2;
        l1 += Mf[(1 * 3 + 0) * CV_ + k] * v0 + Mf[(1 * 3 + 1) * CV_ + k] * v1 + Mf[(1 * 3 + 2) * CV_ + k] * v2;
        l2 += Mf[(2 * 3 + 0) * CV_ + k] * v0 + Mf[(2 * 3 + 1) * CV_ + k] * v1 + Mf[(2 * 3 + 2) * CV_ + k] * v2;
    }
    const float mx = fmaxf(l0, fmaxf(l1, l2));
    const float e0 = __expf(l0 - mx), e1 = __expf(l1 - mx), e2 = __expf(l2 - mx);
    const float inv = 1.0f / (e0 + e1 + e2);
    const float w0 = e0 * inv, w1 = e1 * inv, w2 = e2 * inv;

#pragma unroll 7
    for (int k = 0; k < CV_; ++k) {
        const size_t off = base + (size_t)k * HW_;
        final_out[off] = corr[off] + w0 * cv0[off] + w1 * cv1[off] + w2 * cv2[off];
    }
    w_out[(b * 3 + 0) * HW_ + hw] = w0;
    w_out[(b * 3 + 1) * HW_ + hw] = w1;
    w_out[(b * 3 + 2) * HW_ + hw] = w2;
}

// -------------------------------------------------------------------------
extern "C" void kernel_launch(void* const* d_in, const int* in_sizes, int n_in,
                              void* d_out, int out_size, void* d_ws, size_t ws_size,
                              hipStream_t stream)
{
    const float* feat_l1 = (const float*)d_in[0];
    const float* feat_r1 = (const float*)d_in[1];
    const float* cv0     = (const float*)d_in[2];
    const float* cv1     = (const float*)d_in[3];
    const float* cv2     = (const float*)d_in[4];
    const float* W_f1    = (const float*)d_in[5];
    const float* b_f1    = (const float*)d_in[6];
    const float* W_f2    = (const float*)d_in[7];
    const float* b_f2    = (const float*)d_in[8];
    const float* W_geo   = (const float*)d_in[9];
    const float* b_geo   = (const float*)d_in[10];
    const float* W_sel   = (const float*)d_in[11];
    const float* b_sel   = (const float*)d_in[12];

    char* ws = (char*)d_ws;
    _Float16* f1n = (_Float16*)(ws + F1OFF);
    _Float16* f2n = (_Float16*)(ws + F2OFF);
    float*    Mf  = (float*)(ws + MOFF);
    _Float16* Wh  = (_Float16*)(ws + WOFF);

    float* out       = (float*)d_out;
    float* final_out = out;                 // output 0: final
    float* corr_out  = out + NCORR;         // output 1: init_corr
    float* w_out     = out + 2 * NCORR;     // output 2: w

    wconv_kernel<<<dim3((2 * F_ * C_) / 256), 256, 0, stream>>>(W_f1, W_f2, Wh);
    fold_kernel<<<dim3(1), 512, 0, stream>>>(W_sel, b_sel, W_geo, b_geo, Mf);
    fmap_kernel<<<dim3(W_ / 16, H_, B_ * 2), 128, 0, stream>>>(
        feat_l1, feat_r1, Wh, b_f1, b_f2, f1n, f2n);
    corr_kernel<<<dim3(4, H_, B_), 128, 0, stream>>>(f1n, f2n, corr_out);
    select_kernel<<<dim3((B_ * HW_) / 256), 256, 0, stream>>>(
        cv0, cv1, cv2, Mf, corr_out, final_out, w_out);
}
